// Decoder_GCNMOE_44573170598231
// MI455X (gfx1250) — compile-verified, matrix operand order checked by reading, not run on hardware
//
#include <hip/hip_runtime.h>

// ---------------------------------------------------------------------------
// CDNA5 (gfx1250) WMMA implementation of the GRU+MoE decoder.
// wave32; matrix ops via v_wmma_f32_16x16x32_f16 (f16 in, f32 accumulate).
// Weight tiles are staged in LDS once per workgroup (8 waves share B tiles).
// ---------------------------------------------------------------------------

typedef __attribute__((ext_vector_type(16))) _Float16 v16h;
typedef __attribute__((ext_vector_type(8)))  _Float16 v8h;
typedef __attribute__((ext_vector_type(8)))  float    v8f;

#define DEVI __device__ __forceinline__

constexpr int BS  = 128;
constexpr int T   = 384;
constexpr int D   = 256;
constexpr int H   = 256;
constexpr int NC  = 12;
constexpr int E   = 4;
constexpr int NJ  = 24;
constexpr int NF  = 6;
constexpr int MOE_H = 512;
constexpr int OUTC  = NJ * NF;        // 144
constexpr int NROWS = BS * T;         // 49152, row index n = t*BS + b
constexpr int G3H   = 3 * H;          // 768
constexpr int EMB_IN = D + NC + 1;    // 269

DEVI float sigmoidf_(float x) { return 1.f / (1.f + __expf(-x)); }
DEVI float eluf_(float x)     { return x > 0.f ? x : __expf(x) - 1.f; }

// Per-lane WMMA fragment load for A (row-major, row = M) or B (row-major
// weight row = output col N). K index of element e: (e&7) + 8*(lane>>4)
// + 16*(e>>3)  ->  two contiguous 8-half (16B) loads at k0 and k0+16.
// Works for both global and LDS-resident rows (addrspace inferred).
DEVI v16h load_frag(const _Float16* row, int k0) {
  v8h lo = *reinterpret_cast<const v8h*>(row + k0);
  v8h hi = *reinterpret_cast<const v8h*>(row + k0 + 16);
  v16h r;
#pragma unroll
  for (int i = 0; i < 8; ++i) { r[i] = lo[i]; r[i + 8] = hi[i]; }
  return r;
}

#define WMMA16(A_, B_, C_)                                                    \
  __builtin_amdgcn_wmma_f32_16x16x32_f16(false, (A_), false, (B_), (short)0,  \
                                         (C_), false, false)

// f32 -> f16 conversion (grid-stride)
__global__ void k_f32_to_f16(const float* __restrict__ in,
                             _Float16* __restrict__ out, size_t n) {
  size_t i = (size_t)blockIdx.x * blockDim.x + threadIdx.x;
  size_t stride = (size_t)gridDim.x * blockDim.x;
  for (; i < n; i += stride) out[i] = (_Float16)in[i];
}

// base[b][h] = z[b]·W[h,0:D] + W[h,D+y[b]] + emb_b[h]; wtime[h] = W[h,D+NC]
__global__ void k_embed_base(const float* __restrict__ z,
                             const int* __restrict__ y,
                             const float* __restrict__ emb_w,
                             const float* __restrict__ emb_b,
                             float* __restrict__ base,
                             float* __restrict__ wtime) {
  int idx = blockIdx.x * blockDim.x + threadIdx.x;
  if (idx >= BS * H) return;
  int h = idx & (H - 1);
  int b = idx >> 8;
  const float* wrow = emb_w + (size_t)h * EMB_IN;
  float acc = emb_b[h] + wrow[D + y[b]];
  const float* zr = z + (size_t)b * D;
#pragma unroll 4
  for (int k = 0; k < D; ++k) acc += zr[k] * wrow[k];
  base[idx] = acc;
  if (b == 0) wtime[h] = wrow[D + NC];
}

__global__ void k_build_x(const float* __restrict__ base,
                          const float* __restrict__ wtime,
                          const unsigned char* __restrict__ mask,
                          const int* __restrict__ lengths,
                          _Float16* __restrict__ X /* [T,B,H] */) {
  size_t idx = (size_t)blockIdx.x * blockDim.x + threadIdx.x;
  if (idx >= (size_t)NROWS * H) return;
  int h = (int)(idx & (H - 1));
  int b = (int)((idx >> 8) & (BS - 1));
  int t = (int)(idx >> 15);
  float tm = mask[(size_t)b * T + t] ? ((float)t / ((float)lengths[b] - 1.f))
                                     : 0.f;
  X[idx] = (_Float16)(base[b * H + h] + tm * wtime[h]);
}

// ---------------------------------------------------------------------------
// Generic WMMA GEMM: out[n,o] = act( sum_k A[n,k]*W[o,k] + bias[o] )
// Block = 8 waves = 128 rows x 64 cols. The 64xK weight tile is staged in
// LDS once per block (dynamic smem = 64*K*2 bytes), so weights are fetched
// from L2 once per 128 rows instead of once per 16 rows.
// ---------------------------------------------------------------------------
__global__ void k_gemm_bias_act(const _Float16* __restrict__ A,
                                const _Float16* __restrict__ W,
                                const float* __restrict__ bias,
                                _Float16* __restrict__ out,
                                int N, int K, int O, int act) {
  extern __shared__ _Float16 sB[];  // [64][K]
  int lane = threadIdx.x & 31;
  int waveInBlk = threadIdx.x >> 5;  // 0..7
  int colGroups = O >> 6;
  int rowBlk = blockIdx.x / colGroups;
  int cg     = blockIdx.x % colGroups;
  int obase  = cg * 64;

  // cooperative stage: W rows [obase, obase+64) -> LDS
  int vecs = (64 * K) >> 3;
  for (int idx = threadIdx.x; idx < vecs; idx += blockDim.x) {
    int row = idx / (K >> 3);
    int c8  = (idx % (K >> 3)) << 3;
    *reinterpret_cast<v8h*>(sB + row * K + c8) =
        *reinterpret_cast<const v8h*>(W + (size_t)(obase + row) * K + c8);
  }
  __syncthreads();

  int nbase = rowBlk * 128 + waveInBlk * 16;
  if (nbase >= N) return;
  int laneLo = lane & 15, laneHi = lane >> 4;

  const _Float16* arow = A + (size_t)(nbase + laneLo) * K;
  const _Float16* b0 = sB + (size_t)(0 * 16 + laneLo) * K;
  const _Float16* b1 = sB + (size_t)(1 * 16 + laneLo) * K;
  const _Float16* b2 = sB + (size_t)(2 * 16 + laneLo) * K;
  const _Float16* b3 = sB + (size_t)(3 * 16 + laneLo) * K;

  v8f c0 = {}, c1 = {}, c2 = {}, c3 = {};
  for (int kc = 0; kc < K; kc += 32) {
    int k0 = kc + (laneHi << 3);
    v16h a  = load_frag(arow, k0);
    v16h f0 = load_frag(b0, k0);
    v16h f1 = load_frag(b1, k0);
    v16h f2 = load_frag(b2, k0);
    v16h f3 = load_frag(b3, k0);
    c0 = WMMA16(a, f0, c0);
    c1 = WMMA16(a, f1, c1);
    c2 = WMMA16(a, f2, c2);
    c3 = WMMA16(a, f3, c3);
  }
#pragma unroll
  for (int sub = 0; sub < 4; ++sub) {
    v8f acc = (sub == 0) ? c0 : (sub == 1) ? c1 : (sub == 2) ? c2 : c3;
    int o = obase + sub * 16 + laneLo;
    float bo = bias[o];
#pragma unroll
    for (int r = 0; r < 8; ++r) {
      int n = nbase + r + (laneHi << 3);
      float v = acc[r] + bo;
      if (act) v = eluf_(v);
      out[(size_t)n * O + o] = (_Float16)v;
    }
  }
}

// ---------------------------------------------------------------------------
// One GRU step: gh = Hprev @ Whh^T for r/z/n column triples (3 accumulators),
// fused with gate nonlinearity using precomputed Gi = x@wih^T + bih.
// Block = one 16-col tile, 8 waves cover all 8 batch-row tiles; the 3 gate
// weight tiles (3 x 16 x H = 24 KB) are staged in LDS once per block.
// Grid: 16 blocks x 256 threads.
// ---------------------------------------------------------------------------
__global__ void k_gru_step(const _Float16* __restrict__ Gi /* [N, 3H] */,
                           const _Float16* __restrict__ Hprev /* [B, H] */,
                           const _Float16* __restrict__ Whh /* [3H, H] */,
                           const float* __restrict__ bhh /* [3H] */,
                           _Float16* __restrict__ Xout /* [T,B,H] */, int t) {
  __shared__ _Float16 sW[3 * 16 * H];  // 24 KB
  int lane = threadIdx.x & 31;
  int bt   = threadIdx.x >> 5;  // 0..7 batch-row tile
  int jbase = blockIdx.x * 16;  // 16 col tiles

  // stage Whh rows {g*H + jbase + r : g in 0..2, r in 0..15}
  constexpr int vecs = (48 * H) >> 3;
  for (int idx = threadIdx.x; idx < vecs; idx += 256) {
    int rr = idx / (H >> 3);          // 0..47
    int c8 = (idx % (H >> 3)) << 3;
    int grow = (rr >> 4) * H + jbase + (rr & 15);
    *reinterpret_cast<v8h*>(sW + rr * H + c8) =
        *reinterpret_cast<const v8h*>(Whh + (size_t)grow * H + c8);
  }
  __syncthreads();

  int laneLo = lane & 15, laneHi = lane >> 4;
  int bbase = bt * 16;

  const _Float16* arow = Hprev + (size_t)(bbase + laneLo) * H;
  const _Float16* wr = sW + (size_t)(0 * 16 + laneLo) * H;
  const _Float16* wz = sW + (size_t)(1 * 16 + laneLo) * H;
  const _Float16* wn = sW + (size_t)(2 * 16 + laneLo) * H;

  v8f cr = {}, cz = {}, cn = {};
  for (int kc = 0; kc < H; kc += 32) {
    int k0 = kc + (laneHi << 3);
    v16h a  = load_frag(arow, k0);
    v16h br = load_frag(wr, k0);
    v16h bz = load_frag(wz, k0);
    v16h bn = load_frag(wn, k0);
    cr = WMMA16(a, br, cr);
    cz = WMMA16(a, bz, cz);
    cn = WMMA16(a, bn, cn);
  }

  int j = jbase + laneLo;
  float br_ = bhh[j], bz_ = bhh[H + j], bn_ = bhh[2 * H + j];
#pragma unroll
  for (int r = 0; r < 8; ++r) {
    int b = bbase + r + (laneHi << 3);
    size_t gi = ((size_t)t * BS + b) * G3H + j;
    float i_r = (float)Gi[gi];
    float i_z = (float)Gi[gi + H];
    float i_n = (float)Gi[gi + 2 * H];
    float rg = sigmoidf_(i_r + cr[r] + br_);
    float ug = sigmoidf_(i_z + cz[r] + bz_);
    float ng = tanhf(i_n + rg * (cn[r] + bn_));
    float hp = (float)Hprev[(size_t)b * H + j];
    float hv = (1.f - ug) * ng + ug * hp;
    Xout[((size_t)t * BS + b) * H + j] = (_Float16)hv;
  }
}

// Gate layer 3 (512 -> 4) + softmax. Output width 4 < tile: plain VALU.
__global__ void k_gate2_softmax(const _Float16* __restrict__ G1,
                                const float* __restrict__ g2w /* [4,512] */,
                                const float* __restrict__ g2b,
                                float* __restrict__ gw /* [N,4] */, int N) {
  int n = blockIdx.x * blockDim.x + threadIdx.x;
  if (n >= N) return;
  float a0 = g2b[0], a1 = g2b[1], a2 = g2b[2], a3 = g2b[3];
  const _Float16* row = G1 + (size_t)n * MOE_H;
#pragma unroll 4
  for (int k = 0; k < MOE_H; ++k) {
    float v = (float)row[k];
    a0 += v * g2w[k];
    a1 += v * g2w[MOE_H + k];
    a2 += v * g2w[2 * MOE_H + k];
    a3 += v * g2w[3 * MOE_H + k];
  }
  float m = fmaxf(fmaxf(a0, a1), fmaxf(a2, a3));
  float e0 = __expf(a0 - m), e1 = __expf(a1 - m);
  float e2 = __expf(a2 - m), e3 = __expf(a3 - m);
  float inv = 1.f / (e0 + e1 + e2 + e3);
  gw[(size_t)n * E + 0] = e0 * inv;
  gw[(size_t)n * E + 1] = e1 * inv;
  gw[(size_t)n * E + 2] = e2 * inv;
  gw[(size_t)n * E + 3] = e3 * inv;
}

// ---------------------------------------------------------------------------
// Fused MoE blend: out[n,o] = act( sum_e gw[n,e]*(A@alpha[e]^T + beta[e]) )
// Block = 8 waves = 128 rows x 16 cols. All 4 experts' 16xF weight tiles are
// staged in LDS once per block (dynamic smem = 4*16*F*2 bytes).
// ---------------------------------------------------------------------------
__global__ void k_moe_blend(const _Float16* __restrict__ A,
                            const _Float16* __restrict__ alpha /* [E,O,F] */,
                            const float* __restrict__ beta /* [E,O] */,
                            const float* __restrict__ gw /* [N,E] */,
                            _Float16* __restrict__ out, int N, int F, int O,
                            int act) {
  extern __shared__ _Float16 sB[];  // [E][16][F]
  int lane = threadIdx.x & 31;
  int waveInBlk = threadIdx.x >> 5;
  int colTiles = O >> 4;
  int rowBlk = blockIdx.x / colTiles;
  int ct     = blockIdx.x % colTiles;
  int obase  = ct * 16;

  int vecs = (E * 16 * F) >> 3;
  for (int idx = threadIdx.x; idx < vecs; idx += blockDim.x) {
    int rr = idx / (F >> 3);            // 0..E*16-1: e = rr>>4, r = rr&15
    int c8 = (idx % (F >> 3)) << 3;
    int grow = (rr >> 4) * O + obase + (rr & 15);
    *reinterpret_cast<v8h*>(sB + rr * F + c8) =
        *reinterpret_cast<const v8h*>(alpha + (size_t)grow * F + c8);
  }
  __syncthreads();

  int nbase = rowBlk * 128 + waveInBlk * 16;
  if (nbase >= N) return;
  int laneLo = lane & 15, laneHi = lane >> 4;
  const _Float16* arow = A + (size_t)(nbase + laneLo) * F;
  int o = obase + laneLo;

  float blend[8];
#pragma unroll
  for (int r = 0; r < 8; ++r) blend[r] = 0.f;

  for (int e = 0; e < E; ++e) {
    const _Float16* wrow = sB + (size_t)(e * 16 + laneLo) * F;
    v8f acc = {};
    for (int kc = 0; kc < F; kc += 32) {
      int k0 = kc + (laneHi << 3);
      v16h a = load_frag(arow, k0);
      v16h b = load_frag(wrow, k0);
      acc = WMMA16(a, b, acc);
    }
    float be = beta[(size_t)e * O + o];
#pragma unroll
    for (int r = 0; r < 8; ++r) {
      int n = nbase + r + (laneHi << 3);
      blend[r] += gw[(size_t)n * E + e] * (acc[r] + be);
    }
  }
#pragma unroll
  for (int r = 0; r < 8; ++r) {
    int n = nbase + r + (laneHi << 3);
    float v = blend[r];
    if (act) v = eluf_(v);
    out[(size_t)n * O + o] = (_Float16)v;
  }
}

// Final blend layer (O=144) fused with mask + (B,NJ,NF,T) transpose.
__global__ void k_moe_final(const _Float16* __restrict__ A /* [N, 512] */,
                            const _Float16* __restrict__ alpha /* [E,144,512] */,
                            const float* __restrict__ beta /* [E,144] */,
                            const float* __restrict__ gw /* [N,E] */,
                            const unsigned char* __restrict__ mask,
                            float* __restrict__ outp /* [B,NJ,NF,T] */) {
  constexpr int O = OUTC, F = MOE_H;
  extern __shared__ _Float16 sB[];  // [E][16][F]
  int lane = threadIdx.x & 31;
  int waveInBlk = threadIdx.x >> 5;
  constexpr int colTiles = O / 16;  // 9
  int rowBlk = blockIdx.x / colTiles;
  int ct     = blockIdx.x % colTiles;
  int obase  = ct * 16;

  constexpr int vecs = (E * 16 * F) >> 3;
  for (int idx = threadIdx.x; idx < vecs; idx += 256) {
    int rr = idx / (F >> 3);
    int c8 = (idx % (F >> 3)) << 3;
    int grow = (rr >> 4) * O + obase + (rr & 15);
    *reinterpret_cast<v8h*>(sB + rr * F + c8) =
        *reinterpret_cast<const v8h*>(alpha + (size_t)grow * F + c8);
  }
  __syncthreads();

  int nbase = rowBlk * 128 + waveInBlk * 16;
  if (nbase >= NROWS) return;
  int laneLo = lane & 15, laneHi = lane >> 4;
  const _Float16* arow = A + (size_t)(nbase + laneLo) * F;
  int o = obase + laneLo;

  float blend[8];
#pragma unroll
  for (int r = 0; r < 8; ++r) blend[r] = 0.f;

  for (int e = 0; e < E; ++e) {
    const _Float16* wrow = sB + (size_t)(e * 16 + laneLo) * F;
    v8f acc = {};
    for (int kc = 0; kc < F; kc += 32) {
      int k0 = kc + (laneHi << 3);
      v16h a = load_frag(arow, k0);
      v16h b = load_frag(wrow, k0);
      acc = WMMA16(a, b, acc);
    }
    float be = beta[(size_t)e * O + o];
#pragma unroll
    for (int r = 0; r < 8; ++r) {
      int n = nbase + r + (laneHi << 3);
      blend[r] += gw[(size_t)n * E + e] * (acc[r] + be);
    }
  }
  int j = o / NF, f = o % NF;
#pragma unroll
  for (int r = 0; r < 8; ++r) {
    int n = nbase + r + (laneHi << 3);
    int t = n >> 7;
    int b = n & (BS - 1);
    float v = blend[r];
    if (!mask[(size_t)b * T + t]) v = 0.f;
    outp[(((size_t)b * NJ + j) * NF + f) * T + t] = v;
  }
}

extern "C" void kernel_launch(void* const* d_in, const int* in_sizes, int n_in,
                              void* d_out, int out_size, void* d_ws,
                              size_t ws_size, hipStream_t stream) {
  (void)in_sizes; (void)n_in; (void)out_size; (void)ws_size;
  const float* z       = (const float*)d_in[0];
  const int*   y       = (const int*)d_in[1];
  const unsigned char* mask = (const unsigned char*)d_in[2];
  const int*   lengths = (const int*)d_in[3];
  const float* emb_w   = (const float*)d_in[4];
  const float* emb_b   = (const float*)d_in[5];
  const float* gru_wih = (const float*)d_in[6];
  const float* gru_whh = (const float*)d_in[7];
  const float* gru_bih = (const float*)d_in[8];
  const float* gru_bhh = (const float*)d_in[9];
  const float* g0_w = (const float*)d_in[10];
  const float* g0_b = (const float*)d_in[11];
  const float* g1_w = (const float*)d_in[12];
  const float* g1_b = (const float*)d_in[13];
  const float* g2_w = (const float*)d_in[14];
  const float* g2_b = (const float*)d_in[15];
  const float* alpha0 = (const float*)d_in[16];
  const float* beta0  = (const float*)d_in[17];
  const float* alpha1 = (const float*)d_in[18];
  const float* beta1  = (const float*)d_in[19];
  const float* alpha2 = (const float*)d_in[20];
  const float* beta2  = (const float*)d_in[21];
  float* outp = (float*)d_out;

  char* ws = (char*)d_ws;
  size_t off = 0;
  auto carve = [&](size_t bytes) -> char* {
    off = (off + 255) & ~(size_t)255;
    char* p = ws + off;
    off += bytes;
    return p;
  };
  _Float16* Xa    = (_Float16*)carve((size_t)NROWS * H * 2);
  _Float16* Xb    = (_Float16*)carve((size_t)NROWS * H * 2);
  _Float16* Gi    = (_Float16*)carve((size_t)NROWS * G3H * 2);   // also g0out, h1
  _Float16* g1buf = (_Float16*)carve((size_t)NROWS * MOE_H * 2); // also g1out, h2
  _Float16* wih16 = (_Float16*)carve((size_t)4 * G3H * H * 2);
  _Float16* whh16 = (_Float16*)carve((size_t)4 * G3H * H * 2);
  _Float16* g0w16 = (_Float16*)carve((size_t)MOE_H * H * 2);
  _Float16* g1w16 = (_Float16*)carve((size_t)MOE_H * MOE_H * 2);
  _Float16* a0h   = (_Float16*)carve((size_t)E * MOE_H * H * 2);
  _Float16* a1h   = (_Float16*)carve((size_t)E * MOE_H * MOE_H * 2);
  _Float16* a2h   = (_Float16*)carve((size_t)E * OUTC * MOE_H * 2);
  float*    basef = (float*)carve((size_t)BS * H * 4);
  float*    wtime = (float*)carve((size_t)H * 4);
  _Float16* hzero = (_Float16*)carve((size_t)BS * H * 2);
  float*    gwbuf = (float*)carve((size_t)NROWS * E * 4);

  auto cvt = [&](const float* src, _Float16* dst, size_t n) {
    int blocks = (int)((n + 255) / 256);
    if (blocks > 2048) blocks = 2048;
    k_f32_to_f16<<<blocks, 256, 0, stream>>>(src, dst, n);
  };
  cvt(gru_wih, wih16, (size_t)4 * G3H * H);
  cvt(gru_whh, whh16, (size_t)4 * G3H * H);
  cvt(g0_w,   g0w16, (size_t)MOE_H * H);
  cvt(g1_w,   g1w16, (size_t)MOE_H * MOE_H);
  cvt(alpha0, a0h,   (size_t)E * MOE_H * H);
  cvt(alpha1, a1h,   (size_t)E * MOE_H * MOE_H);
  cvt(alpha2, a2h,   (size_t)E * OUTC * MOE_H);

  hipMemsetAsync(hzero, 0, (size_t)BS * H * 2, stream);

  k_embed_base<<<(BS * H) / 256, 256, 0, stream>>>(z, y, emb_w, emb_b, basef,
                                                   wtime);
  k_build_x<<<(NROWS * H) / 256, 256, 0, stream>>>(basef, wtime, mask, lengths,
                                                   Xa);

  // 4-layer GRU: hoisted input GEMM + 384 fused recurrent steps per layer.
  _Float16* Xcur = Xa;
  _Float16* Xnext = Xb;
  for (int l = 0; l < 4; ++l) {
    int blocks = (NROWS / 128) * (G3H / 64);        // 384*12 = 4608
    size_t smem = (size_t)64 * H * 2;               // 32 KB (K=H)
    k_gemm_bias_act<<<blocks, 256, smem, stream>>>(
        Xcur, wih16 + (size_t)l * G3H * H, gru_bih + (size_t)l * G3H, Gi,
        NROWS, H, G3H, 0);
    for (int t = 0; t < T; ++t) {
      const _Float16* hprev =
          (t == 0) ? hzero : (Xnext + (size_t)(t - 1) * BS * H);
      k_gru_step<<<H / 16, 256, 0, stream>>>(Gi, hprev,
                                             whh16 + (size_t)l * G3H * H,
                                             gru_bhh + (size_t)l * G3H, Xnext,
                                             t);
    }
    _Float16* tmp = Xcur; Xcur = Xnext; Xnext = tmp;
  }
  _Float16* zf = Xcur;

  // Gate MLP
  _Float16* g0out = Gi;  // reuse Gi region
  {
    int blocks = (NROWS / 128) * (MOE_H / 64);
    size_t smem = (size_t)64 * H * 2;  // K=256 -> 32 KB
    k_gemm_bias_act<<<blocks, 256, smem, stream>>>(zf, g0w16, g0_b, g0out,
                                                   NROWS, H, MOE_H, 1);
  }
  _Float16* g1out = g1buf;
  {
    int blocks = (NROWS / 128) * (MOE_H / 64);
    size_t smem = (size_t)64 * MOE_H * 2;  // K=512 -> 64 KB
    k_gemm_bias_act<<<blocks, 256, smem, stream>>>(g0out, g1w16, g1_b, g1out,
                                                   NROWS, MOE_H, MOE_H, 1);
  }
  k_gate2_softmax<<<NROWS / 256, 256, 0, stream>>>(g1out, g2_w, g2_b, gwbuf,
                                                   NROWS);

  // Fused MoE blend layers
  _Float16* h1 = Gi;     // g0out dead after g1 GEMM
  {
    int blocks = (NROWS / 128) * (MOE_H / 16);
    size_t smem = (size_t)E * 16 * H * 2;  // F=256 -> 32 KB
    k_moe_blend<<<blocks, 256, smem, stream>>>(zf, a0h, beta0, gwbuf, h1,
                                               NROWS, H, MOE_H, 1);
  }
  _Float16* h2 = g1buf;  // g1out dead after gate2
  {
    int blocks = (NROWS / 128) * (MOE_H / 16);
    size_t smem = (size_t)E * 16 * MOE_H * 2;  // F=512 -> 64 KB
    k_moe_blend<<<blocks, 256, smem, stream>>>(h1, a1h, beta1, gwbuf, h2,
                                               NROWS, MOE_H, MOE_H, 1);
  }
  {
    int blocks = (NROWS / 128) * (OUTC / 16);
    size_t smem = (size_t)E * 16 * MOE_H * 2;  // 64 KB
    k_moe_final<<<blocks, 256, smem, stream>>>(h2, a2h, beta2, gwbuf, mask,
                                               outp);
  }
}
